// Cross_View_45646912422477
// MI455X (gfx1250) — matrix operand order repaired and checked by reading, not compiled, not run on hardware
//
#include <hip/hip_runtime.h>

#define H 128
#define N_P 100000
#define N_A 50000
#define N_C 10000
#define N_R 30000
#define NE  500000
#define NMASK 50000

typedef __attribute__((ext_vector_type(16))) _Float16 v16h;
typedef __attribute__((ext_vector_type(8)))  float    v8f;

__device__ __forceinline__ float warp_sum(float v) {
#pragma unroll
    for (int off = 16; off > 0; off >>= 1) v += __shfl_xor(v, off, 32);
    return v;
}

// float atomic-max via sign-split int ordering (init target to -inf)
__device__ __forceinline__ void atomic_max_f32(float* addr, float val) {
    if (val >= 0.0f) atomicMax((int*)addr, __float_as_int(val));
    else             atomicMin((unsigned int*)addr, __float_as_uint(val));
}

// ---------------- Phase 1: Y = prelu(X @ W + b) via v_wmma_f32_16x16x32_f16 ----
// One wave (32 lanes) computes a 16x16 output tile; K=128 -> 4 WMMA steps.
// Fragments (wave32, 16-bit A 16x32; B 32x16; f32 C/D): A/B in contiguous k = 16*half + s order.
__global__ __launch_bounds__(32)
void gemm_prelu_wmma(const float* __restrict__ X, const float* __restrict__ W,
                     const float* __restrict__ bias, const float* __restrict__ act_arr,
                     int act_idx, float* __restrict__ Y, int nrows) {
    const int lane = threadIdx.x & 31;
    const int ml   = lane & 15;   // N index for B/C/D, M index for A rows
    const int g    = lane >> 4;   // lane-half
    const int rowBase = blockIdx.x * 16;
    const int colBase = blockIdx.y * 16;

    v8f c = {};
    const float* xrow = X + (size_t)(rowBase + ml) * H;

#pragma unroll
    for (int kc = 0; kc < 4; ++kc) {
        const int kb = kc * 32;
        v16h a, b;
#pragma unroll
        for (int v = 0; v < 8; ++v) {
            // A 16x32 f16: lanes 0-15 K{0..15}, lanes 16-31 K{16..31}
            int ka = kb + 2 * v + 16 * g;
            float2 xa = *(const float2*)(xrow + ka);
            a[2 * v]     = (_Float16)xa.x;
            a[2 * v + 1] = (_Float16)xa.y;
            // B 32x16 f16: lane n=ml holds column, VGPR v holds K=2v(+16 for upper half)
            int kcol = kb + 2 * v + 16 * g;
            b[2 * v]     = (_Float16)W[(size_t)kcol * H + colBase + ml];
            b[2 * v + 1] = (_Float16)W[(size_t)(kcol + 1) * H + colBase + ml];
        }
        c = __builtin_amdgcn_wmma_f32_16x16x32_f16(false, a, false, b, (short)0, c,
                                                   false, false);
    }

    const float act  = act_arr[act_idx];
    const float bcol = bias[colBase + ml];
#pragma unroll
    for (int v = 0; v < 8; ++v) {           // C/D: VGPR v -> row v (+8 upper half)
        int row = rowBase + v + 8 * g;
        float y = c[v] + bcol;
        y = (y >= 0.0f) ? y : act * y;
        Y[(size_t)row * H + colBase + ml] = y;
    }
}

// ---------------- GAT building blocks --------------------------------------
__global__ void dot_per_node(const float* __restrict__ X, const float* __restrict__ att,
                             float* __restrict__ out, int n) {
    int w    = (blockIdx.x * blockDim.x + threadIdx.x) >> 5;
    int lane = threadIdx.x & 31;
    if (w >= n) return;
    float4 x = *(const float4*)(X + (size_t)w * H + lane * 4);
    float4 a = *(const float4*)(att + lane * 4);
    float s = x.x * a.x + x.y * a.y + x.z * a.z + x.w * a.w;
    s = warp_sum(s);
    if (lane == 0) out[w] = s;
}

__global__ void fill_f32(float* __restrict__ p, float v, int n) {
    int i = blockIdx.x * blockDim.x + threadIdx.x;
    if (i < n) p[i] = v;
}

__global__ void edge_logits_max(const int* __restrict__ ei, const float* __restrict__ a_src,
                                const float* __restrict__ a_dst, float* __restrict__ ew,
                                float* __restrict__ mx, int ne) {
    int e = blockIdx.x * blockDim.x + threadIdx.x;
    if (e >= ne) return;
    int s = ei[e], d = ei[ne + e];
    float a = a_src[s] + a_dst[d];
    a = (a >= 0.0f) ? a : 0.01f * a;        // leaky_relu(0.01)
    ew[e] = a;
    atomic_max_f32(&mx[d], a);
}

__global__ void edge_exp_sum(const int* __restrict__ ei, float* __restrict__ ew,
                             const float* __restrict__ mx, float* __restrict__ ss, int ne) {
    int e = blockIdx.x * blockDim.x + threadIdx.x;
    if (e >= ne) return;
    int d = ei[ne + e];
    float w = __expf(ew[e] - mx[d]);
    ew[e] = w;
    atomicAdd(&ss[d], w);
}

// One wave per edge: gather 128-f32 src row (float4/lane), scatter-add into dst row.
__global__ void edge_aggregate(const int* __restrict__ ei, const float* __restrict__ ew,
                               const float* __restrict__ ss, const float* __restrict__ xsrc,
                               float* __restrict__ G, int ne) {
    int e    = (blockIdx.x * blockDim.x + threadIdx.x) >> 5;
    int lane = threadIdx.x & 31;
    if (e >= ne) return;
    int s = ei[e], d = ei[ne + e];
    float coef = ew[e] / ss[d];
    float4 v = *(const float4*)(xsrc + (size_t)s * H + lane * 4);
    float* gd = G + (size_t)d * H + lane * 4;
    atomicAdd(gd + 0, v.x * coef);
    atomicAdd(gd + 1, v.y * coef);
    atomicAdd(gd + 2, v.z * coef);
    atomicAdd(gd + 3, v.w * coef);
}

// ---------------- BatchNorm ------------------------------------------------
__global__ __launch_bounds__(H)
void bn_stats(const float* __restrict__ X, float* __restrict__ sums, int n) {
    int c = threadIdx.x;                    // column 0..127 (coalesced)
    float s = 0.0f, s2 = 0.0f;
    for (int r = blockIdx.x; r < n; r += gridDim.x) {
        float x = X[(size_t)r * H + c];
        s += x; s2 += x * x;
    }
    atomicAdd(&sums[c], s);
    atomicAdd(&sums[H + c], s2);
}

__global__ void bn_prelu_apply(const float* __restrict__ X, const float* __restrict__ sums,
                               const float* __restrict__ gamma, const float* __restrict__ beta,
                               const float* __restrict__ prelu_arr, int pidx,
                               float* __restrict__ Y, int n) {
    int i = blockIdx.x * blockDim.x + threadIdx.x;
    if (i >= n * H) return;
    int c = i & (H - 1);
    float inv_n = 1.0f / (float)n;
    float mu  = sums[c] * inv_n;
    float var = sums[H + c] * inv_n - mu * mu;
    float y = (X[i] - mu) * rsqrtf(var + 1e-5f) * gamma[c] + beta[c];
    float pw = prelu_arr[pidx];
    Y[i] = (y >= 0.0f) ? y : pw * y;
}

// masked nodes' dst-attention logit becomes dot(enc_mask_token, att_dst)
__global__ void mask_adst(float* __restrict__ a_dst, const int* __restrict__ mask,
                          const float* __restrict__ token, const float* __restrict__ att,
                          int nm) {
    int j = blockIdx.x * blockDim.x + threadIdx.x;
    if (j >= nm) return;
    float s = 0.0f;
#pragma unroll 8
    for (int c = 0; c < H; ++c) s += token[c] * att[c];
    a_dst[mask[j]] = s;
}

// One wave per masked node: bn+prelu on the fly, cosine-embedding loss accumulate.
__global__ void loss_kernel(const float* __restrict__ G, const float* __restrict__ sums,
                            const float* __restrict__ gamma, const float* __restrict__ beta,
                            const float* __restrict__ prelu_arr, int pidx,
                            const float* __restrict__ Sa, const float* __restrict__ Sb,
                            const int* __restrict__ mask, int nm, int nbn,
                            float* __restrict__ out) {
    int j    = (blockIdx.x * blockDim.x + threadIdx.x) >> 5;
    int lane = threadIdx.x & 31;
    if (j >= nm) return;
    int p = mask[j];
    float inv_n = 1.0f / (float)nbn;
    float pw = prelu_arr[pidx];
    float dee = 0.0f, dhh = 0.0f, deh = 0.0f;
#pragma unroll
    for (int k = 0; k < 4; ++k) {
        int c = lane * 4 + k;
        float mu  = sums[c] * inv_n;
        float var = sums[H + c] * inv_n - mu * mu;
        float x  = G[(size_t)p * H + c];
        float e2 = (x - mu) * rsqrtf(var + 1e-5f) * gamma[c] + beta[c];
        e2 = (e2 >= 0.0f) ? e2 : pw * e2;
        float h1 = Sa[(size_t)p * H + c] + Sb[(size_t)p * H + c];
        dee += e2 * e2; dhh += h1 * h1; deh += e2 * h1;
    }
    dee = warp_sum(dee); dhh = warp_sum(dhh); deh = warp_sum(deh);
    if (lane == 0) {
        float ne   = fmaxf(sqrtf(dee), 1e-12f);
        float nh   = fmaxf(sqrtf(dhh), 1e-12f);
        float cosv = deh / (ne * nh);
        atomicAdd(out, (1.0f - cosv) / (float)nm);
    }
}

// ---------------------------------------------------------------------------
extern "C" void kernel_launch(void* const* d_in, const int* in_sizes, int n_in,
                              void* d_out, int out_size, void* d_ws, size_t ws_size,
                              hipStream_t stream) {
    (void)in_sizes; (void)n_in; (void)out_size; (void)ws_size;
    const float* xP = (const float*)d_in[0];
    const float* xA = (const float*)d_in[1];
    const float* xC = (const float*)d_in[2];
    const float* xR = (const float*)d_in[3];
    const float* wP = (const float*)d_in[4];  const float* bP = (const float*)d_in[5];
    const float* wA = (const float*)d_in[6];  const float* bA = (const float*)d_in[7];
    const float* wC = (const float*)d_in[8];  const float* bC = (const float*)d_in[9];
    const float* wR = (const float*)d_in[10]; const float* bR = (const float*)d_in[11];
    const float* act_w  = (const float*)d_in[12];
    const float* token  = (const float*)d_in[13];
    const float* mp_as  = (const float*)d_in[14];
    const float* mp_ad  = (const float*)d_in[15];
    /* mp_bias d_in[16] dropped: per-column const is annihilated by BN mean-subtract */
    const float* mp_g   = (const float*)d_in[17];
    const float* mp_b   = (const float*)d_in[18];
    const float* mp_pr  = (const float*)d_in[19];
    const float* sc_as  = (const float*)d_in[20];
    const float* sc_ad  = (const float*)d_in[21];
    /* sc_bias d_in[22] dropped for the same reason */
    const float* sc_g   = (const float*)d_in[23];
    const float* sc_b   = (const float*)d_in[24];
    const float* sc_pr  = (const float*)d_in[25];
    const int* ei_PA = (const int*)d_in[26];
    const int* ei_PC = (const int*)d_in[27];
    const int* ei_PR = (const int*)d_in[28];
    const int* ei_AP = (const int*)d_in[29];
    const int* ei_CP = (const int*)d_in[30];
    const int* ei_RP = (const int*)d_in[31];
    const int* m1 = (const int*)d_in[32];
    const int* m2 = (const int*)d_in[33];
    const int* m3 = (const int*)d_in[34];

    // ---- workspace carve (~306 MB) ----
    char* wsp = (char*)d_ws;
    auto carve = [&](size_t nf) { float* p = (float*)wsp; wsp += nf * sizeof(float); return p; };
    float* hP = carve((size_t)N_P * H);
    float* hA = carve((size_t)N_A * H);
    float* hC = carve((size_t)N_C * H);
    float* hR = carve((size_t)N_R * H);
    float* S0 = carve((size_t)N_P * H);
    float* S1 = carve((size_t)N_P * H);
    float* S2 = carve((size_t)N_P * H);
    float* G  = carve((size_t)N_P * H);     // shared GAT accumulator (max n_dst)
    float* a_src  = carve(N_P);
    float* a_dst  = carve(N_P);
    float* mvec   = carve(N_P);
    float* svec   = carve(N_P);
    float* ew     = carve(NE);
    float* bnsums = carve(2 * H);

    auto run_gemm = [&](const float* X, const float* W, const float* B, int aidx,
                        float* Y, int n) {
        dim3 grid(n / 16, H / 16);
        gemm_prelu_wmma<<<grid, dim3(32), 0, stream>>>(X, W, B, act_w, aidx, Y, n);
    };
    auto run_dot = [&](const float* X, const float* att, float* o, int n) {
        dot_per_node<<<(n + 7) / 8, 256, 0, stream>>>(X, att, o, n);
    };
    auto run_edges = [&](const float* xsrc, int n_dst, const int* ei, float* Gout) {
        fill_f32<<<(n_dst + 255) / 256, 256, 0, stream>>>(mvec, -__builtin_inff(), n_dst);
        hipMemsetAsync(svec, 0, (size_t)n_dst * sizeof(float), stream);
        edge_logits_max<<<(NE + 255) / 256, 256, 0, stream>>>(ei, a_src, a_dst, ew, mvec, NE);
        edge_exp_sum<<<(NE + 255) / 256, 256, 0, stream>>>(ei, ew, mvec, svec, NE);
        hipMemsetAsync(Gout, 0, (size_t)n_dst * H * sizeof(float), stream);
        edge_aggregate<<<(NE * 32 + 255) / 256, 256, 0, stream>>>(ei, ew, svec, xsrc, Gout, NE);
    };
    auto run_bn_stats = [&](const float* X, int n) {
        hipMemsetAsync(bnsums, 0, 2 * H * sizeof(float), stream);
        bn_stats<<<512, H, 0, stream>>>(X, bnsums, n);
    };

    // ---- Phase 1: dense transforms (WMMA) ----
    run_gemm(xP, wP, bP, 0, hP, N_P);
    run_gemm(xA, wA, bA, 1, hA, N_A);
    run_gemm(xC, wC, bC, 2, hC, N_C);
    run_gemm(xR, wR, bR, 3, hR, N_R);

    // ---- Phase 2: metapath GATs P->{A,C,R}, overwrite h_dst ----
    float* mp_h[3]      = {hA, hC, hR};
    int    mp_n[3]      = {N_A, N_C, N_R};
    const int* mp_ei[3] = {ei_PA, ei_PC, ei_PR};
    for (int i = 0; i < 3; ++i) {
        run_dot(hP, mp_as + i * H, a_src, N_P);
        run_dot(mp_h[i], mp_ad + i * H, a_dst, mp_n[i]);
        run_edges(hP, mp_n[i], mp_ei[i], G);
        run_bn_stats(G, mp_n[i]);
        int tot = mp_n[i] * H;
        bn_prelu_apply<<<(tot + 255) / 256, 256, 0, stream>>>(
            G, bnsums, mp_g + i * H, mp_b + i * H, mp_pr, i, mp_h[i], mp_n[i]);
    }

    // ---- Phase 3: cached schema GATs S_i = schema(i, hP) ----
    float* sc_h[3]      = {hA, hC, hR};
    int    sc_n[3]      = {N_A, N_C, N_R};
    const int* sc_ei[3] = {ei_AP, ei_CP, ei_RP};
    float* Sbuf[3]      = {S0, S1, S2};
    for (int i = 0; i < 3; ++i) {
        run_dot(sc_h[i], sc_as + i * H, a_src, sc_n[i]);
        run_dot(hP, sc_ad + i * H, a_dst, N_P);
        run_edges(sc_h[i], N_P, sc_ei[i], G);
        run_bn_stats(G, N_P);
        bn_prelu_apply<<<(N_P * H + 255) / 256, 256, 0, stream>>>(
            G, bnsums, sc_g + i * H, sc_b + i * H, sc_pr, i, Sbuf[i], N_P);
    }

    // ---- Phase 4: three masked passes -> scalar loss ----
    hipMemsetAsync(d_out, 0, sizeof(float), stream);
    int        lp_t[3]  = {1, 2, 0};
    int        lp_c0[3] = {0, 0, 1};
    int        lp_c1[3] = {2, 1, 2};
    const int* lp_m[3]  = {m1, m2, m3};
    for (int l = 0; l < 3; ++l) {
        int t = lp_t[l];
        run_dot(sc_h[t], sc_as + t * H, a_src, sc_n[t]);
        run_dot(hP, sc_ad + t * H, a_dst, N_P);
        mask_adst<<<(NMASK + 255) / 256, 256, 0, stream>>>(a_dst, lp_m[l], token,
                                                           sc_ad + t * H, NMASK);
        run_edges(sc_h[t], N_P, sc_ei[t], G);
        run_bn_stats(G, N_P);
        loss_kernel<<<(NMASK * 32 + 255) / 256, 256, 0, stream>>>(
            G, bnsums, sc_g + t * H, sc_b + t * H, sc_pr, t,
            Sbuf[lp_c0[l]], Sbuf[lp_c1[l]], lp_m[l], NMASK, N_P, (float*)d_out);
    }
}
